// MultiHeadAttention_54984171323694
// MI455X (gfx1250) — compile-verified, matrix operand order checked by reading, not run on hardware
//
#include <hip/hip_runtime.h>

typedef __bf16 bf16;
typedef __attribute__((ext_vector_type(4)))  bf16  v4bf;
typedef __attribute__((ext_vector_type(8)))  bf16  v8bf;
typedef __attribute__((ext_vector_type(16))) bf16  v16bf;
typedef __attribute__((ext_vector_type(8)))  float v8f;
typedef __attribute__((ext_vector_type(4)))  unsigned int v4u;
typedef __attribute__((ext_vector_type(4)))  int  v4i;
typedef __attribute__((ext_vector_type(8)))  int  v8i;

#define WMMA_BF16(a, b, c) \
  __builtin_amdgcn_wmma_f32_16x16x32_bf16(false, (a), false, (b), (short)0, (c), false, false)

static __device__ inline v16bf cat8(v8bf lo, v8bf hi) {
  return __builtin_shufflevector(lo, hi, 0,1,2,3,4,5,6,7,8,9,10,11,12,13,14,15);
}

// ---------------------------------------------------------------------------
// Tensor Data Mover: 2D tile (bf16) global -> LDS, with optional LDS row pad.
// D# layout per CDNA5 ISA 8.3/8.4. Groups 2/3 zero (<=2D tensor).
//   pad_interval_code: pad after 2^(c+1) DWORDs ; pad_amount_code: (c+1) DWORDs
// ---------------------------------------------------------------------------
static __device__ inline void tdm_load_2d_bf16(
    unsigned lds_off, const void* gaddr,
    unsigned tile_d0, unsigned tile_d1,     // elems (d0 contiguous)
    unsigned stride_d0,                     // elems between rows in memory
    unsigned pad_interval_code, unsigned pad_amount_code, bool pad_en) {
  unsigned long long ga = (unsigned long long)(size_t)gaddr;
  v4u g0;
  g0.x = 1u;                                            // count = 1 valid D#
  g0.y = lds_off;                                       // lds_addr (bytes)
  g0.z = (unsigned)(ga & 0xffffffffu);                  // global_addr[31:0]
  g0.w = (unsigned)((ga >> 32) & 0x01ffffffu) | (2u << 30);  // addr[56:32] | type=2
  v8i g1;
  unsigned w0 = (1u << 16);                             // data_size = 2 bytes
  if (pad_en) w0 |= (1u << 20) | (pad_interval_code << 22) | (pad_amount_code << 25);
  g1[0] = (int)w0;
  g1[1] = (int)((tile_d0 & 0xffffu) << 16);             // tensor_dim0 (== tile_d0) low16
  g1[2] = (int)((tile_d1 & 0xffffu) << 16);             // tensor_dim0 hi16=0 | tensor_dim1 low16
  g1[3] = (int)((tile_d0 & 0xffffu) << 16);             // tensor_dim1 hi16=0 | tile_dim0
  g1[4] = (int)(tile_d1 & 0xffffu);                     // tile_dim1 | tile_dim2=0
  g1[5] = (int)stride_d0;                               // tensor_dim0_stride low32
  g1[6] = 0;                                            // stride0 hi | stride1 lo
  g1[7] = 0;
  v4i z4 = {0, 0, 0, 0};
#if __has_include(<hip/amd_detail/amd_gfx1250_TDM.h>)
  v8i z8 = {0, 0, 0, 0, 0, 0, 0, 0};
  __builtin_amdgcn_tensor_load_to_lds(g0, g1, z4, z4, z8, 0);
#else
  __builtin_amdgcn_tensor_load_to_lds(g0, g1, z4, z4, 0);
#endif
}

// ---------------------------------------------------------------------------
// fp32 -> bf16 conversion (vectorized: float4 in, 4x bf16 out)
// ---------------------------------------------------------------------------
__global__ void cvt_f32_bf16(const float* __restrict__ s, bf16* __restrict__ d, int n4) {
  int i = blockIdx.x * 256 + threadIdx.x;
  if (i >= n4) return;
  float4 v = ((const float4*)s)[i];
  v4bf o;
  o.x = (bf16)v.x; o.y = (bf16)v.y; o.z = (bf16)v.z; o.w = (bf16)v.w;
  ((v4bf*)d)[i] = o;
}

// ---------------------------------------------------------------------------
// QKV GEMM: [8192 x 768] @ [768 x 2304] -> scatter into Q/K/V [48][2048][64]
// A tile staged by TDM (wave 0), B tile transposed by all threads.
// ---------------------------------------------------------------------------
__global__ __launch_bounds__(256)
void gemm_qkv(const bf16* __restrict__ A, const bf16* __restrict__ Bm,
              bf16* __restrict__ qo, bf16* __restrict__ ko, bf16* __restrict__ vo) {
  constexpr int LDA = 768, LDB = 2304;
  __shared__ bf16 As[128][40];   // 32 elem rows + 8 pad (TDM pad: 16 DW interval, 4 DW pad)
  __shared__ bf16 Bt[64][40];    // B tile transposed: [n][k]
  const int tid = threadIdx.x, lane = tid & 31;
  const int wid = tid >> 5;
  const int wm = (wid & 3) * 32, wn = (wid >> 2) * 32;
  const int m0 = blockIdx.y * 128, n0 = blockIdx.x * 64;
  const int lr = lane & 15, kb = (lane >> 4) * 8;
  const unsigned lds_as = (unsigned)(size_t)&As[0][0];

  v8f acc[2][2] = {};
  for (int kk = 0; kk < 768; kk += 32) {
    __syncthreads();
    if (wid == 0)   // A tile via Tensor Data Mover: 128 rows x 32 k, stride 768
      tdm_load_2d_bf16(lds_as, A + (size_t)m0 * LDA + kk, 32, 128, LDA, 3, 3, true);
    {   // B tile transposed scatter: 32 k x 64 n, 8 elems per thread
      int krow = tid >> 3, c0 = (tid & 7) * 8;
      v8bf v = *(const v8bf*)(Bm + (size_t)(kk + krow) * LDB + n0 + c0);
      #pragma unroll
      for (int j = 0; j < 8; ++j) Bt[c0 + j][krow] = v[j];
      if (kk + 32 < 768)
        __builtin_prefetch(Bm + (size_t)(kk + 32 + krow) * LDB + n0 + c0, 0, 1);
    }
    if (wid == 0) __builtin_amdgcn_s_wait_tensorcnt((short)0);
    __syncthreads();

    v16bf af[2], bfr[2];
    #pragma unroll
    for (int i = 0; i < 2; ++i) {
      int r = wm + 16 * i + lr;
      af[i] = cat8(*(const v8bf*)&As[r][kb], *(const v8bf*)&As[r][kb + 16]);
    }
    #pragma unroll
    for (int j = 0; j < 2; ++j) {
      int c = wn + 16 * j + lr;
      bfr[j] = cat8(*(const v8bf*)&Bt[c][kb], *(const v8bf*)&Bt[c][kb + 16]);
    }
    #pragma unroll
    for (int i = 0; i < 2; ++i)
      #pragma unroll
      for (int j = 0; j < 2; ++j)
        acc[i][j] = WMMA_BF16(af[i], bfr[j], acc[i][j]);
  }

  // Epilogue: C layout row = r + 8*(lane>>4), col = lane&15 per 16x16 tile.
  #pragma unroll
  for (int i = 0; i < 2; ++i) {
    #pragma unroll
    for (int j = 0; j < 2; ++j) {
      #pragma unroll
      for (int r = 0; r < 8; ++r) {
        int rg = m0 + wm + 16 * i + r + 8 * (lane >> 4);
        int cg = n0 + wn + 16 * j + lr;
        int b = rg >> 11, n = rg & 2047;
        int which = cg / 768, rem = cg % 768;
        int h = rem >> 6, hd = rem & 63;
        bf16* dst = (which == 0) ? qo : (which == 1 ? ko : vo);
        dst[(((size_t)b * 12 + h) * 2048 + n) * 64 + hd] = (bf16)acc[i][j][r];
      }
    }
  }
}

// ---------------------------------------------------------------------------
// Flash attention: grid (48 head-batches, 32 q-tiles); 128 thr = 4 waves,
// each wave owns 16 query rows. KV tiles of 64, online softmax.
// K tile staged by TDM (wave 0); V transposed manually by all threads.
// ---------------------------------------------------------------------------
__global__ __launch_bounds__(128)
void flash_attn(const bf16* __restrict__ qbuf, const bf16* __restrict__ kbuf,
                const bf16* __restrict__ vbuf, bf16* __restrict__ obuf) {
  constexpr float SCALE = 0.125f;   // 64^-0.5
  __shared__ bf16 Kt[64][72];       // [key][hd]; TDM pad: 32 DW interval, 4 DW pad
  __shared__ bf16 Vt[64][72];       // [hd][key] (transposed)
  __shared__ bf16 Pt[4][16][72];    // per-wave P tile
  const int bh = blockIdx.x;
  const int tid = threadIdx.x, lane = tid & 31, wid = tid >> 5;
  const int lr = lane & 15, kb = (lane >> 4) * 8, hh = lane >> 4;
  const int q0 = blockIdx.y * 64 + wid * 16;
  const size_t base = (size_t)bh * 2048 * 64;
  const unsigned lds_kt = (unsigned)(size_t)&Kt[0][0];

  // Q fragments in registers for the whole KV loop
  v16bf aq[2];
  {
    const bf16* qp = qbuf + base + (size_t)(q0 + lr) * 64;
    #pragma unroll
    for (int s = 0; s < 2; ++s)
      aq[s] = cat8(*(const v8bf*)(qp + 32 * s + kb),
                   *(const v8bf*)(qp + 32 * s + kb + 16));
  }

  v8f acc_o[4] = {};
  float mrow[8], lrow[8];
  #pragma unroll
  for (int r = 0; r < 8; ++r) { mrow[r] = -1e30f; lrow[r] = 0.f; }

  for (int kt0 = 0; kt0 < 2048; kt0 += 64) {
    __syncthreads();
    if (wid == 0)   // K tile via Tensor Data Mover: 64 keys x 64 hd, stride 64
      tdm_load_2d_bf16(lds_kt, kbuf + base + (size_t)kt0 * 64, 64, 64, 64, 4, 3, true);
    {   // V tile transposed to [hd][key] (TDM cannot transpose)
      int key = tid >> 1, c0 = (tid & 1) * 32;
      const bf16* g = vbuf + base + (size_t)(kt0 + key) * 64 + c0;
      #pragma unroll
      for (int c = 0; c < 4; ++c) {
        v8bf v = *(const v8bf*)(g + 8 * c);
        #pragma unroll
        for (int j = 0; j < 8; ++j) Vt[c0 + 8 * c + j][key] = v[j];
      }
      if (kt0 + 64 < 2048) {
        __builtin_prefetch(vbuf + base + (size_t)(kt0 + 64 + key) * 64 + c0, 0, 1);
        __builtin_prefetch(kbuf + base + (size_t)(kt0 + 64 + key) * 64 + c0, 0, 1);
      }
    }
    if (wid == 0) __builtin_amdgcn_s_wait_tensorcnt((short)0);
    __syncthreads();

    // S = Q K^T : 4 key sub-tiles x 2 k-chunks = 8 WMMA
    v8f sc[4] = {};
    #pragma unroll
    for (int t = 0; t < 4; ++t) {
      int col = 16 * t + lr;
      #pragma unroll
      for (int s = 0; s < 2; ++s) {
        v16bf bk = cat8(*(const v8bf*)&Kt[col][32 * s + kb],
                        *(const v8bf*)&Kt[col][32 * s + kb + 16]);
        sc[t] = WMMA_BF16(aq[s], bk, sc[t]);
      }
    }
    #pragma unroll
    for (int t = 0; t < 4; ++t) sc[t] = sc[t] * SCALE;

    // Online softmax: row reductions stay inside 16-lane halves (C layout).
    #pragma unroll
    for (int r = 0; r < 8; ++r) {
      float m = fmaxf(fmaxf(sc[0][r], sc[1][r]), fmaxf(sc[2][r], sc[3][r]));
      #pragma unroll
      for (int mask = 1; mask <= 8; mask <<= 1)
        m = fmaxf(m, __shfl_xor(m, mask, 32));
      float mn = fmaxf(mrow[r], m);
      float corr = __expf(mrow[r] - mn);
      float ls = 0.f;
      #pragma unroll
      for (int t = 0; t < 4; ++t) {
        float p = __expf(sc[t][r] - mn);
        sc[t][r] = p;
        ls += p;
      }
      #pragma unroll
      for (int mask = 1; mask <= 8; mask <<= 1)
        ls += __shfl_xor(ls, mask, 32);
      lrow[r] = lrow[r] * corr + ls;
      mrow[r] = mn;
      #pragma unroll
      for (int t = 0; t < 4; ++t) acc_o[t][r] = acc_o[t][r] * corr;
    }

    // P -> per-wave LDS tile (reshape C-layout -> A-fragment layout)
    #pragma unroll
    for (int t = 0; t < 4; ++t)
      #pragma unroll
      for (int r = 0; r < 8; ++r)
        Pt[wid][r + 8 * hh][16 * t + lr] = (bf16)sc[t][r];

    v16bf ap[2];
    #pragma unroll
    for (int s = 0; s < 2; ++s)
      ap[s] = cat8(*(const v8bf*)&Pt[wid][lr][32 * s + kb],
                   *(const v8bf*)&Pt[wid][lr][32 * s + kb + 16]);

    // O += P V : 4 hd sub-tiles x 2 k-chunks = 8 WMMA
    #pragma unroll
    for (int t = 0; t < 4; ++t) {
      int col = 16 * t + lr;
      #pragma unroll
      for (int s = 0; s < 2; ++s) {
        v16bf bv = cat8(*(const v8bf*)&Vt[col][32 * s + kb],
                        *(const v8bf*)&Vt[col][32 * s + kb + 16]);
        acc_o[t] = WMMA_BF16(ap[s], bv, acc_o[t]);
      }
    }
  }

  // Normalize and write attention output as [B,N,(h d)] bf16 for proj GEMM.
  const int b = bh / 12, h = bh % 12;
  #pragma unroll
  for (int r = 0; r < 8; ++r) {
    float inv = 1.0f / lrow[r];
    int n = blockIdx.y * 64 + wid * 16 + r + 8 * hh;
    #pragma unroll
    for (int t = 0; t < 4; ++t) {
      int hd = 16 * t + lr;
      obuf[((size_t)b * 2048 + n) * 768 + h * 64 + hd] = (bf16)(acc_o[t][r] * inv);
    }
  }
}

// ---------------------------------------------------------------------------
// Proj GEMM: [8192 x 768] @ [768 x 768] + bias -> fp32 out
// ---------------------------------------------------------------------------
__global__ __launch_bounds__(256)
void gemm_proj(const bf16* __restrict__ A, const bf16* __restrict__ Bm,
               const float* __restrict__ bias, float* __restrict__ out) {
  constexpr int LDA = 768, LDB = 768;
  __shared__ bf16 As[128][40];
  __shared__ bf16 Bt[64][40];
  const int tid = threadIdx.x, lane = tid & 31;
  const int wid = tid >> 5;
  const int wm = (wid & 3) * 32, wn = (wid >> 2) * 32;
  const int m0 = blockIdx.y * 128, n0 = blockIdx.x * 64;
  const int lr = lane & 15, kb = (lane >> 4) * 8;
  const unsigned lds_as = (unsigned)(size_t)&As[0][0];

  v8f acc[2][2] = {};
  for (int kk = 0; kk < 768; kk += 32) {
    __syncthreads();
    if (wid == 0)
      tdm_load_2d_bf16(lds_as, A + (size_t)m0 * LDA + kk, 32, 128, LDA, 3, 3, true);
    {
      int krow = tid >> 3, c0 = (tid & 7) * 8;
      v8bf v = *(const v8bf*)(Bm + (size_t)(kk + krow) * LDB + n0 + c0);
      #pragma unroll
      for (int j = 0; j < 8; ++j) Bt[c0 + j][krow] = v[j];
      if (kk + 32 < 768)
        __builtin_prefetch(Bm + (size_t)(kk + 32 + krow) * LDB + n0 + c0, 0, 1);
    }
    if (wid == 0) __builtin_amdgcn_s_wait_tensorcnt((short)0);
    __syncthreads();

    v16bf af[2], bfr[2];
    #pragma unroll
    for (int i = 0; i < 2; ++i) {
      int r = wm + 16 * i + lr;
      af[i] = cat8(*(const v8bf*)&As[r][kb], *(const v8bf*)&As[r][kb + 16]);
    }
    #pragma unroll
    for (int j = 0; j < 2; ++j) {
      int c = wn + 16 * j + lr;
      bfr[j] = cat8(*(const v8bf*)&Bt[c][kb], *(const v8bf*)&Bt[c][kb + 16]);
    }
    #pragma unroll
    for (int i = 0; i < 2; ++i)
      #pragma unroll
      for (int j = 0; j < 2; ++j)
        acc[i][j] = WMMA_BF16(af[i], bfr[j], acc[i][j]);
  }

  #pragma unroll
  for (int i = 0; i < 2; ++i) {
    #pragma unroll
    for (int j = 0; j < 2; ++j) {
      #pragma unroll
      for (int r = 0; r < 8; ++r) {
        int rg = m0 + wm + 16 * i + r + 8 * (lane >> 4);
        int cg = n0 + wn + 16 * j + lr;
        out[(size_t)rg * 768 + cg] = acc[i][j][r] + bias[cg];
      }
    }
  }
}

// ---------------------------------------------------------------------------
extern "C" void kernel_launch(void* const* d_in, const int* in_sizes, int n_in,
                              void* d_out, int out_size, void* d_ws, size_t ws_size,
                              hipStream_t stream) {
  const float* x      = (const float*)d_in[0];
  const float* w_qkv  = (const float*)d_in[1];
  const float* w_proj = (const float*)d_in[2];
  const float* b_proj = (const float*)d_in[3];
  float* out = (float*)d_out;

  char* w = (char*)d_ws;
  size_t off = 0;
  auto alloc = [&](size_t bytes) -> void* {
    void* p = w + off;
    off += (bytes + 255) & ~(size_t)255;
    return p;
  };
  bf16* xb     = (bf16*)alloc((size_t)8192 * 768 * 2);
  bf16* wqkvb  = (bf16*)alloc((size_t)768 * 2304 * 2);
  bf16* wprojb = (bf16*)alloc((size_t)768 * 768 * 2);
  bf16* qb     = (bf16*)alloc((size_t)48 * 2048 * 64 * 2);
  bf16* kbuf   = (bf16*)alloc((size_t)48 * 2048 * 64 * 2);
  bf16* vbuf   = (bf16*)alloc((size_t)48 * 2048 * 64 * 2);
  bf16* attnb  = (bf16*)alloc((size_t)8192 * 768 * 2);
  (void)ws_size; (void)in_sizes; (void)n_in; (void)out_size;

  auto cvt = [&](const float* s, bf16* d, int n) {
    int n4 = n / 4;
    cvt_f32_bf16<<<(n4 + 255) / 256, 256, 0, stream>>>(s, d, n4);
  };
  cvt(x,      xb,     8192 * 768);
  cvt(w_qkv,  wqkvb,  768 * 2304);
  cvt(w_proj, wprojb, 768 * 768);

  gemm_qkv<<<dim3(2304 / 64, 8192 / 128), 256, 0, stream>>>(xb, wqkvb, qb, kbuf, vbuf);
  flash_attn<<<dim3(48, 2048 / 64), 128, 0, stream>>>(qb, kbuf, vbuf, attnb);
  gemm_proj<<<dim3(768 / 64, 8192 / 128), 256, 0, stream>>>(attnb, wprojb, b_proj, out);
}
